// SoftTargetInfoNCE_91285234909639
// MI455X (gfx1250) — matrix operand order repaired and checked by reading, not compiled, not run on hardware
//
#include <hip/hip_runtime.h>

// ---------------------------------------------------------------------------
// SoftTargetInfoNCE, fused:  loss = mean_n( logsumexp_m(q·t^T) - (q·t^T)[n,n] )
//   q = logits/T - log(noise_probs),  T = 1
// Split-bf16 (hi+lo) GEMM on v_wmma_f32_16x16x32_bf16 fused with a streaming
// logsumexp (base-2; raw v_exp_f32).
//   - 2 row tiles x 1/2 column range per wave: halves L2 B-fragment traffic
//     at constant wave count (1024); 4 independent WMMA chains for the XDL.
//   - 3-buffer rotating register pipeline (3x unrolled, static slots): no
//     modulo-scheduling copies, rolling-pointer addressing.
//   - fmaxf (lowers to v_max_num, dual-issued), raw exp2.
//   - per-row partial (m, s, diag) to workspace; finalize merges halves and
//     does a deterministic tree reduction (no atomics).
// ---------------------------------------------------------------------------

typedef __bf16 bf16_t;
typedef bf16_t v16bf __attribute__((ext_vector_type(16)));
typedef float  v8f   __attribute__((ext_vector_type(8)));

#define DDIM   64
#define LOG2E  1.4426950408889634f
#define MLN2   0.6931471805599453f

__device__ __forceinline__ unsigned short f2bf(float x) {
  unsigned int u = __float_as_uint(x);
  unsigned int r = u + 0x7FFFu + ((u >> 16) & 1u);
  return (unsigned short)(r >> 16);
}
__device__ __forceinline__ float bf2f(unsigned short h) {
  return __uint_as_float(((unsigned int)h) << 16);
}

__device__ __forceinline__ v8f wmma_bf16(v16bf a, v16bf b, v8f c) {
  return __builtin_amdgcn_wmma_f32_16x16x32_bf16(false, a, false, b,
                                                 (short)0, c, false, false);
}

// raw v_exp_f32 (args always <= 0 here; flush-to-zero tail is exactly right)
__device__ __forceinline__ float fast_exp2(float x) {
  return __builtin_amdgcn_exp2f(x);
}

struct Frag4 { v16bf h0, h1, l0, l1; };   // {hi,lo} x {K-chunk0, K-chunk1}

__device__ __forceinline__ Frag4 load_frag_p(const unsigned short* ph,
                                             const unsigned short* pl) {
  Frag4 f;
  f.h0 = *(const v16bf*)(ph);
  f.h1 = *(const v16bf*)(ph + 512);
  f.l0 = *(const v16bf*)(pl);
  f.l1 = *(const v16bf*)(pl + 512);
  return f;
}

__device__ __forceinline__ Frag4 load_frag(const unsigned short* hi,
                                           const unsigned short* lo,
                                           int u, int lane) {
  size_t off = (size_t)u * 1024 + (size_t)lane * 16;
  return load_frag_p(hi + off, lo + off);
}

// 12 WMMAs, 4 independent depth-3 chains (2 row tiles x 2 K-chunks)
__device__ __forceinline__ void mm12(const Frag4& a0, const Frag4& a1,
                                     const Frag4& b,
                                     v8f& c00, v8f& c01, v8f& c10, v8f& c11) {
  c00 = wmma_bf16(a0.h0, b.h0, c00);
  c01 = wmma_bf16(a0.h1, b.h1, c01);
  c10 = wmma_bf16(a1.h0, b.h0, c10);
  c11 = wmma_bf16(a1.h1, b.h1, c11);
  c00 = wmma_bf16(a0.h0, b.l0, c00);
  c01 = wmma_bf16(a0.h1, b.l1, c01);
  c10 = wmma_bf16(a1.h0, b.l0, c10);
  c11 = wmma_bf16(a1.h1, b.l1, c11);
  c00 = wmma_bf16(a0.l0, b.h0, c00);
  c01 = wmma_bf16(a0.l1, b.h1, c01);
  c10 = wmma_bf16(a1.l0, b.h0, c10);
  c11 = wmma_bf16(a1.l1, b.h1, c11);
}

// streaming lse update: add, max, 2 sub, 2 raw exp2, fma per element
__device__ __forceinline__ void lse_update(const v8f& acc0, const v8f& acc1,
                                           float m[8], float s[8]) {
#pragma unroll
  for (int r = 0; r < 8; ++r) {
    float c  = acc0[r] + acc1[r];          // scores * log2(e)
    float nm = fmaxf(m[r], c);
    float e1 = fast_exp2(m[r] - nm);
    float e2 = fast_exp2(c - nm);
    s[r] = __builtin_fmaf(s[r], e1, e2);
    m[r] = nm;
  }
}

// one non-pipelined tile (cold path: epilogue / diagonal)
__device__ __forceinline__ void tile_step(const Frag4& a0, const Frag4& a1,
                                          const Frag4& b,
                                          float m0[8], float s0[8],
                                          float m1[8], float s1[8]) {
  v8f p00 = {}, p01 = {}, p10 = {}, p11 = {};
  mm12(a0, a1, b, p00, p01, p10, p11);
  lse_update(p00, p01, m0, s0);
  lse_update(p10, p11, m1, s1);
}

// ---------------------------------------------------------------------------
// Prep A: q = (logits - log(noise_probs)) * log2(e), split bf16 hi/lo, stored
// in WMMA 16-bit A-fragment order (ISA 7.12.2):
//   lane L: row = 16*tile + (L&15), element i -> K = 8*(L>>4) + i + 8*(i>>3)
// ---------------------------------------------------------------------------
__global__ __launch_bounds__(128) void prep_q_kernel(
    const float* __restrict__ logits, const float* __restrict__ nprobs,
    unsigned short* __restrict__ qhi, unsigned short* __restrict__ qlo,
    int ntiles) {
  int idx   = blockIdx.x * blockDim.x + threadIdx.x;
  int lane  = idx & 31;
  int chunk = (idx >> 5) & 1;
  int tile  = idx >> 6;
  if (tile >= ntiles) return;
  int row  = tile * 16 + (lane & 15);
  int half = lane >> 4;
  size_t base = ((size_t)(tile * 2 + chunk) * 32 + lane) * 16;
#pragma unroll
  for (int i = 0; i < 16; ++i) {
    int kloc = 8 * half + i + ((i >> 3) << 3);
    int d    = chunk * 32 + kloc;
    float x  = (logits[(size_t)row * DDIM + d] - __logf(nprobs[d])) * LOG2E;
    unsigned short h = f2bf(x);
    qhi[base + i] = h;
    qlo[base + i] = f2bf(x - bf2f(h));
  }
}

// ---------------------------------------------------------------------------
// Prep B: targets split bf16 hi/lo in WMMA 16-bit B-fragment order:
//   lane L: col = 16*tile + (L&15), element i -> K = 8*(L>>4) + i + 8*(i>>3)
// ---------------------------------------------------------------------------
__global__ __launch_bounds__(128) void prep_t_kernel(
    const float* __restrict__ targets,
    unsigned short* __restrict__ thi, unsigned short* __restrict__ tlo,
    int mtiles) {
  int idx   = blockIdx.x * blockDim.x + threadIdx.x;
  int lane  = idx & 31;
  int chunk = (idx >> 5) & 1;
  int tile  = idx >> 6;
  if (tile >= mtiles) return;
  int col  = tile * 16 + (lane & 15);
  int half = lane >> 4;
  size_t base = ((size_t)(tile * 2 + chunk) * 32 + lane) * 16;
#pragma unroll
  for (int i = 0; i < 16; ++i) {
    int d = chunk * 32 + 8 * half + i + ((i >> 3) << 3);
    float x = targets[(size_t)col * DDIM + d];
    unsigned short h = f2bf(x);
    thi[base + i] = h;
    tlo[base + i] = f2bf(x - bf2f(h));
  }
}

// ---------------------------------------------------------------------------
// Fused GEMM + streaming logsumexp.
// Wave W = (g, h): row tiles {2g, 2g+1}, column-tile range h of 2.
// D-fragment: lane L, vgpr r -> M = r + 8*(L>>4), N = L&15. Each lane owns
// columns {16*u + (L&15)}: softmax state is per-lane; shfl merge at end, then
// per-row partial (m, s, dg) written to workspace (finalize merges halves).
// ---------------------------------------------------------------------------
__global__ __launch_bounds__(128) void score_lse_kernel(
    const unsigned short* __restrict__ qhi, const unsigned short* __restrict__ qlo,
    const unsigned short* __restrict__ thi, const unsigned short* __restrict__ tlo,
    float* __restrict__ pm, float* __restrict__ ps, float* __restrict__ pdg,
    int ntiles, int mtiles, int nrows) {
  int lane = threadIdx.x & 31;
  int wave = threadIdx.x >> 5;
  int W = blockIdx.x * 4 + wave;     // wave-uniform -> EXEC all 1s in body
  int ngroups = (ntiles + 1) >> 1;
  int g = W >> 1;
  int h = W & 1;
  if (g >= ngroups) return;
  int rt0 = 2 * g;
  int rt1 = (2 * g + 1 < ntiles) ? 2 * g + 1 : rt0;
  int mh = mtiles >> 1;
  int mstart = h ? mh : 0;
  int mend   = h ? mtiles : mh;
  int count  = mend - mstart;

  Frag4 a0 = load_frag(qhi, qlo, rt0, lane);
  Frag4 a1 = load_frag(qhi, qlo, rt1, lane);

  float m0[8], s0[8], m1[8], s1[8];
#pragma unroll
  for (int r = 0; r < 8; ++r) {
    m0[r] = -1e30f; s0[r] = 0.f;
    m1[r] = -1e30f; s1[r] = 0.f;
  }

  int u = mstart;
  if (count >= 6) {
    // 3-buffer rotation, 3x unrolled: static slots, prefetch distance 3,
    // rolling lane pointers (one 64-bit add per array per refill)
    const unsigned short* th = thi + (size_t)lane * 16 + (size_t)mstart * 1024;
    const unsigned short* tl = tlo + (size_t)lane * 16 + (size_t)mstart * 1024;
    Frag4 b0 = load_frag_p(th,        tl);
    Frag4 b1 = load_frag_p(th + 1024, tl + 1024);
    Frag4 b2 = load_frag_p(th + 2048, tl + 2048);
    const unsigned short* rh = th + 3072;   // next tile to load
    const unsigned short* rl = tl + 3072;

    for (; u + 5 < mend; u += 3) {
      v8f p00 = {}, p01 = {}, p10 = {}, p11 = {};
      mm12(a0, a1, b0, p00, p01, p10, p11);              // tile u
      b0 = load_frag_p(rh, rl); rh += 1024; rl += 1024;
      lse_update(p00, p01, m0, s0);
      lse_update(p10, p11, m1, s1);

      v8f q00 = {}, q01 = {}, q10 = {}, q11 = {};
      mm12(a0, a1, b1, q00, q01, q10, q11);              // tile u+1
      b1 = load_frag_p(rh, rl); rh += 1024; rl += 1024;
      lse_update(q00, q01, m0, s0);
      lse_update(q10, q11, m1, s1);

      v8f t00 = {}, t01 = {}, t10 = {}, t11 = {};
      mm12(a0, a1, b2, t00, t01, t10, t11);              // tile u+2
      b2 = load_frag_p(rh, rl); rh += 1024; rl += 1024;
      lse_update(t00, t01, m0, s0);
      lse_update(t10, t11, m1, s1);
    }
    // slots still hold tiles u, u+1, u+2 (loaded but unused); the generic
    // tail below simply reloads them -- cold path, 3..5 iterations.
  }
  for (; u < mend; ++u) {               // tail (and whole range if count < 6)
    Frag4 bx = load_frag(thi, tlo, u, lane);
    tile_step(a0, a1, bx, m0, s0, m1, s1);
  }

  int colSel = lane & 15;
  int half   = lane >> 4;

  // peeled diagonal: row tile rt's diagonal lives in column tile rt; only the
  // wave whose column range contains it extracts (others contribute 0)
  float dg0[8], dg1[8];
#pragma unroll
  for (int r = 0; r < 8; ++r) { dg0[r] = 0.f; dg1[r] = 0.f; }
  if (rt0 >= mstart && rt0 < mend) {
    Frag4 bd = load_frag(thi, tlo, rt0, lane);
    v8f p0 = {}, p1 = {}, d2 = {}, d3 = {};
    mm12(a0, a1, bd, p0, p1, d2, d3);    // chains 0/1 are rt0's scores
#pragma unroll
    for (int r = 0; r < 8; ++r)
      if (colSel == r + 8 * half) dg0[r] = p0[r] + p1[r];
  }
  if (rt1 != rt0 && rt1 >= mstart && rt1 < mend) {
    Frag4 bd = load_frag(thi, tlo, rt1, lane);
    v8f p0 = {}, p1 = {}, d2 = {}, d3 = {};
    mm12(a0, a1, bd, d2, d3, p0, p1);    // chains 2/3 are rt1's scores
#pragma unroll
    for (int r = 0; r < 8; ++r)
      if (colSel == r + 8 * half) dg1[r] = p0[r] + p1[r];
  }

  // merge the 16 per-lane column partitions within each half-wave
#pragma unroll
  for (int mask = 1; mask < 16; mask <<= 1) {
#pragma unroll
    for (int r = 0; r < 8; ++r) {
      float om = __shfl_xor(m0[r], mask, 32);
      float os = __shfl_xor(s0[r], mask, 32);
      float od = __shfl_xor(dg0[r], mask, 32);
      float nm = fmaxf(m0[r], om);
      s0[r] = __builtin_fmaf(s0[r], fast_exp2(m0[r] - nm),
                             os * fast_exp2(om - nm));
      m0[r] = nm;
      dg0[r] += od;

      om = __shfl_xor(m1[r], mask, 32);
      os = __shfl_xor(s1[r], mask, 32);
      od = __shfl_xor(dg1[r], mask, 32);
      nm = fmaxf(m1[r], om);
      s1[r] = __builtin_fmaf(s1[r], fast_exp2(m1[r] - nm),
                             os * fast_exp2(om - nm));
      m1[r] = nm;
      dg1[r] += od;
    }
  }

  // write per-row partial state: lane with (L&15)==r writes row r + 8*half
  size_t hb = (size_t)h * nrows;
#pragma unroll
  for (int r = 0; r < 8; ++r) {
    if (colSel == r) {
      size_t row0 = (size_t)rt0 * 16 + r + 8 * half;
      pm [hb + row0] = m0[r];
      ps [hb + row0] = s0[r];
      pdg[hb + row0] = dg0[r];
      size_t row1 = (size_t)rt1 * 16 + r + 8 * half;
      pm [hb + row1] = m1[r];
      ps [hb + row1] = s1[r];
      pdg[hb + row1] = dg1[r];
    }
  }
}

// merge column halves per row, then deterministic tree reduction -> mean
__global__ __launch_bounds__(256) void finalize_kernel(
    const float* __restrict__ pm, const float* __restrict__ ps,
    const float* __restrict__ pdg, int nrows, float* __restrict__ out) {
  __shared__ float sm[256];
  float v = 0.f;
  for (int n = threadIdx.x; n < nrows; n += 256) {
    float a  = pm[n],        b  = pm[nrows + n];
    float sa = ps[n],        sb = ps[nrows + n];
    float mm = fmaxf(a, b);
    float sr = __builtin_fmaf(sa, fast_exp2(a - mm), sb * fast_exp2(b - mm));
    v += mm + __log2f(sr) - (pdg[n] + pdg[nrows + n]);
  }
  sm[threadIdx.x] = v;
  __syncthreads();
  for (int stride = 128; stride > 0; stride >>= 1) {
    if ((int)threadIdx.x < stride) sm[threadIdx.x] += sm[threadIdx.x + stride];
    __syncthreads();
  }
  if (threadIdx.x == 0) out[0] = sm[0] * (MLN2 / (float)nrows);
}

extern "C" void kernel_launch(void* const* d_in, const int* in_sizes, int n_in,
                              void* d_out, int out_size, void* d_ws, size_t ws_size,
                              hipStream_t stream) {
  const float* logits  = (const float*)d_in[0];
  const float* targets = (const float*)d_in[1];
  const float* nprobs  = (const float*)d_in[2];
  const int D = in_sizes[2];            // 64
  const int N = in_sizes[0] / D;        // 16384
  const int M = in_sizes[1] / D;        // 16384
  const int ntiles = N / 16;
  const int mtiles = M / 16;
  const int nrows  = ntiles * 16;

  unsigned short* qhi = (unsigned short*)d_ws;
  unsigned short* qlo = qhi + (size_t)N * DDIM;
  unsigned short* thi = qlo + (size_t)N * DDIM;
  unsigned short* tlo = thi + (size_t)M * DDIM;
  float* pm  = (float*)(tlo + (size_t)M * DDIM);   // [2][nrows]
  float* ps  = pm  + 2 * (size_t)nrows;            // [2][nrows]
  float* pdg = ps  + 2 * (size_t)nrows;            // [2][nrows]

  prep_q_kernel<<<(ntiles * 64 + 127) / 128, 128, 0, stream>>>(
      logits, nprobs, qhi, qlo, ntiles);
  prep_t_kernel<<<(mtiles * 64 + 127) / 128, 128, 0, stream>>>(
      targets, thi, tlo, mtiles);

  int ngroups = (ntiles + 1) / 2;
  int nwaves  = ngroups * 2;
  score_lse_kernel<<<(nwaves + 3) / 4, 128, 0, stream>>>(
      qhi, qlo, thi, tlo, pm, ps, pdg, ntiles, mtiles, nrows);
  finalize_kernel<<<1, 256, 0, stream>>>(pm, ps, pdg, nrows, (float*)d_out);
}
